// Net_stacked_modified_3839700762819
// MI455X (gfx1250) — matrix operand order repaired and checked by reading, not compiled, not run on hardware
//
#include <hip/hip_runtime.h>

// ---------------------------------------------------------------------------
// Problem constants (match the reference)
// ---------------------------------------------------------------------------
#define B_ROWS 8192
#define DDIM   256
#define HDIM   266
#define TSTEPS 20
#define HPAD_N 272   // H padded to multiple of 16 (output/N padding)
#define HPAD_K 288   // H padded to multiple of 32 (reduction/K padding)
#define BN_EPS 1e-5f
#define NT     4     // N-tiles (16 cols each) per wave in the GEMM

typedef __bf16 v4bf  __attribute__((ext_vector_type(4)));
typedef __bf16 v8bf  __attribute__((ext_vector_type(8)));
typedef __bf16 v16bf __attribute__((ext_vector_type(16)));
typedef float  v8f   __attribute__((ext_vector_type(8)));

__device__ __forceinline__ __bf16 f2bf(float f) {
    unsigned int u = __builtin_bit_cast(unsigned int, f);
    unsigned int r = u + 0x7FFFu + ((u >> 16) & 1u);   // round-to-nearest-even
    unsigned short s = (unsigned short)(r >> 16);
    return __builtin_bit_cast(__bf16, s);
}

// ---------------------------------------------------------------------------
// Weight conversion: src f32 [T x N x K] row-major  ->  dst bf16 [T x Kpad x Npad]
// (transposed so GEMM B-fragment rows are contiguous in N), zero padded.
// ---------------------------------------------------------------------------
__global__ void convert_w_kernel(const float* __restrict__ src, __bf16* __restrict__ dst,
                                 int N, int K, int Npad, int Kpad, int Tn)
{
    long long idx   = (long long)blockIdx.x * blockDim.x + threadIdx.x;
    long long total = (long long)Tn * Kpad * Npad;
    if (idx >= total) return;
    int per = Kpad * Npad;
    int t   = (int)(idx / per);
    int rem = (int)(idx % per);
    int k = rem / Npad, n = rem % Npad;
    float v = (k < K && n < N) ? src[(size_t)t * N * K + (size_t)n * K + k] : 0.0f;
    dst[idx] = f2bf(v);
}

// ---------------------------------------------------------------------------
// Init: xcur(f32) and xbf(bf16) copies of input x
// ---------------------------------------------------------------------------
__global__ void init_x_kernel(const float* __restrict__ x, float* __restrict__ xcur,
                              __bf16* __restrict__ xbf)
{
    int idx = blockIdx.x * blockDim.x + threadIdx.x;   // B*D threads
    float v = x[idx];
    xcur[idx] = v;
    xbf[idx]  = f2bf(v);
}

// ---------------------------------------------------------------------------
// GEMM: out[M x Ntiles*16] = A(bf16)[M x K] @ Bt(bf16)[K x Ntiles*16] + bias
// Bt is the transposed, padded weight (row k holds N contiguous values).
// 4 waves per block; each wave computes one 16-row M tile x NT (=4) N-tiles,
// reusing one A fragment across 4 WMMAs per K-chunk on 4 independent
// accumulator chains. All hot-loop values are scalars (no local arrays) so
// the pointers keep global address-space inference (global_load_b128, not
// flat_load) and the fragments live in distinct registers. Out-of-range N
// tiles are clamped to the last valid tile (computed but not stored). K and
// N fully padded: no bounds checks on loads; bias guarded by Nact.
// ---------------------------------------------------------------------------
__global__ __launch_bounds__(128) void gemm_bf16_kernel(
    const __bf16* __restrict__ A, int lda,
    const __bf16* __restrict__ Bt, int ldb,
    const float* __restrict__ bias, int Nact,
    float* __restrict__ out, int ldo,
    int kChunks, int Ntiles)
{
    const int wave = threadIdx.x >> 5;
    const int lane = threadIdx.x & 31;
    const int m0 = (blockIdx.x * 4 + wave) * 16;
    const int lo = lane & 15;
    const int hi = lane >> 4;

    const int tbase = blockIdx.y * NT;
    int t0 = tbase + 0; if (t0 >= Ntiles) t0 = Ntiles - 1;
    int t1 = tbase + 1; if (t1 >= Ntiles) t1 = Ntiles - 1;
    int t2 = tbase + 2; if (t2 >= Ntiles) t2 = Ntiles - 1;
    int t3 = tbase + 3; if (t3 >= Ntiles) t3 = Ntiles - 1;
    const int n00 = t0 * 16, n01 = t1 * 16, n02 = t2 * 16, n03 = t3 * 16;

    v8f c0 = {}, c1 = {}, c2 = {}, c3 = {};

    // A fragment (16-bit A layout): lane = row m0+lo; half group select = hi
    const __bf16* arow = A + (size_t)(m0 + lo) * lda + hi * 8;
    // B fragments: lane = column n0+lo; elements over k (table order, half select = hi)
    const __bf16* p0 = Bt + (size_t)(hi * 8) * ldb + n00 + lo;
    const __bf16* p1 = Bt + (size_t)(hi * 8) * ldb + n01 + lo;
    const __bf16* p2 = Bt + (size_t)(hi * 8) * ldb + n02 + lo;
    const __bf16* p3 = Bt + (size_t)(hi * 8) * ldb + n03 + lo;
    const size_t bstep = (size_t)32 * ldb;

    for (int kc = 0; kc < kChunks; ++kc) {
        v8bf alo = *(const v8bf*)(arow);        // K = k0 + hi*8 + [0..7]
        v8bf ahi = *(const v8bf*)(arow + 16);   // K = k0 + 16 + hi*8 + [0..7]
        v16bf b0, b1, b2, b3;                   // N = n0j+lo, K same order as a
#pragma unroll
        for (int s = 0; s < 16; ++s) {
            const size_t ko = (size_t)(s < 8 ? s : s + 8) * ldb;
            b0[s] = p0[ko]; b1[s] = p1[ko]; b2[s] = p2[ko]; b3[s] = p3[ko];
        }

        v16bf a = __builtin_shufflevector(alo, ahi,
                    0,1,2,3,4,5,6,7,8,9,10,11,12,13,14,15);
        c0 = __builtin_amdgcn_wmma_f32_16x16x32_bf16(false, a, false, b0, (short)0, c0, false, false);
        c1 = __builtin_amdgcn_wmma_f32_16x16x32_bf16(false, a, false, b1, (short)0, c1, false, false);
        c2 = __builtin_amdgcn_wmma_f32_16x16x32_bf16(false, a, false, b2, (short)0, c2, false, false);
        c3 = __builtin_amdgcn_wmma_f32_16x16x32_bf16(false, a, false, b3, (short)0, c3, false, false);

        arow += 32;
        p0 += bstep; p1 += bstep; p2 += bstep; p3 += bstep;
    }

    const size_t obase = (size_t)(m0 + 8 * hi) * ldo;
#pragma unroll
    for (int j = 0; j < NT; ++j) {
        if (tbase + j < Ntiles) {
            const int n = (j == 0 ? n00 : j == 1 ? n01 : j == 2 ? n02 : n03) + lo;
            const v8f cj = (j == 0 ? c0 : j == 1 ? c1 : j == 2 ? c2 : c3);
            const float bv = (bias && n < Nact) ? bias[n] : 0.0f;
            float* orow = out + obase + n;
#pragma unroll
            for (int r = 0; r < 8; ++r)
                orow[(size_t)r * ldo] = cj[r] + bv;
        }
    }
}

// ---------------------------------------------------------------------------
// BatchNorm statistics -> per-feature scale/shift (deterministic tree reduce)
// scale[f] = g[f]*rsqrt(var+eps), shift[f] = be[f] - mean*scale[f]
// Features >= Hact get scale=shift=0 (zeroes padded activation columns).
// blockDim = (32, 8); grid = Hpad/32 blocks (or fewer for small Hact).
// ---------------------------------------------------------------------------
__global__ __launch_bounds__(256) void stats_kernel(
    const float* __restrict__ y, int ldy, int rows,
    const float* __restrict__ g, const float* __restrict__ be,
    int Hact, float* __restrict__ scale, float* __restrict__ shift)
{
    const int fx = threadIdx.x;           // feature within 32-block
    const int ry = threadIdx.y;           // row-strider 0..7
    const int f  = blockIdx.x * 32 + fx;

    float s = 0.0f, s2 = 0.0f;
    if (f < Hact) {
        for (int r = ry; r < rows; r += 8) {
            float v = y[(size_t)r * ldy + f];
            s  += v;
            s2 += v * v;
        }
    }
    __shared__ float sb[8][32];
    __shared__ float s2b[8][32];
    sb[ry][fx] = s; s2b[ry][fx] = s2;
    __syncthreads();
    if (ry == 0) {
        for (int j = 1; j < 8; ++j) { s += sb[j][fx]; s2 += s2b[j][fx]; }
        if (f < Hact) {
            float inv = 1.0f / (float)rows;
            float m   = s * inv;
            float var = s2 * inv - m * m;
            float sc  = g[f] * rsqrtf(var + BN_EPS);
            scale[f] = sc;
            shift[f] = be[f] - m * sc;
        } else {
            scale[f] = 0.0f;
            shift[f] = 0.0f;
        }
    }
}

// ---------------------------------------------------------------------------
// BN affine + ReLU + cast to bf16 activation buffer [B x HPAD_K]
// Vectorized 4 elements/thread (272 and 288 both multiples of 4).
// ---------------------------------------------------------------------------
__global__ void bnrelu_kernel(const float* __restrict__ y, int ldy,
                              const float* __restrict__ scale,
                              const float* __restrict__ shift,
                              __bf16* __restrict__ h)
{
    int idx = blockIdx.x * blockDim.x + threadIdx.x;   // B * (HPAD_K/4) threads
    int c4 = (idx % (HPAD_K / 4)) * 4;
    int r  =  idx / (HPAD_K / 4);
    v4bf o = {};
    if (c4 < HPAD_N) {   // columns beyond Hact have scale=shift=0 -> relu(0)=0
        const float4 yv = *(const float4*)(y + (size_t)r * ldy + c4);
        o[0] = f2bf(fmaxf(0.0f, yv.x * scale[c4 + 0] + shift[c4 + 0]));
        o[1] = f2bf(fmaxf(0.0f, yv.y * scale[c4 + 1] + shift[c4 + 1]));
        o[2] = f2bf(fmaxf(0.0f, yv.z * scale[c4 + 2] + shift[c4 + 2]));
        o[3] = f2bf(fmaxf(0.0f, yv.w * scale[c4 + 3] + shift[c4 + 3]));
    }
    *(v4bf*)(h + (size_t)r * HPAD_K + c4) = o;
}

// ---------------------------------------------------------------------------
// v0 finalize: v[r] = relu(scale[0]*y[r][0] + shift[0])
// ---------------------------------------------------------------------------
__global__ void v0_finalize_kernel(const float* __restrict__ y, int ldy,
                                   const float* __restrict__ scale,
                                   const float* __restrict__ shift,
                                   float* __restrict__ v)
{
    int r = blockIdx.x * blockDim.x + threadIdx.x;
    v[r] = fmaxf(0.0f, y[(size_t)r * ldy] * scale[0] + shift[0]);
}

// ---------------------------------------------------------------------------
// Per-step update: row reductions (deterministic LDS tree) + x/v update
// alpha = -grad (lambda=1); noise = sqrt(ht)*xi (sigma=1)
// v += -||grad||^2 * ht + <grad, noise>;  x += -grad*ht + noise
// ---------------------------------------------------------------------------
__global__ __launch_bounds__(256) void update_kernel(
    float* __restrict__ xcur, __bf16* __restrict__ xbf,
    float* __restrict__ v, const float* __restrict__ grad,
    const float* __restrict__ xi_t, const float* __restrict__ timegrid, int t)
{
    const int r = blockIdx.x;
    const int d = threadIdx.x;
    const float ht  = timegrid[t + 1] - timegrid[t];
    const float sqh = sqrtf(ht);

    const size_t off = (size_t)r * DDIM + d;
    const float g     = grad[off];
    const float noise = sqh * xi_t[off];

    __shared__ float sf[256];
    __shared__ float sd[256];
    sf[d] = g * g;
    sd[d] = g * noise;
    __syncthreads();
#pragma unroll
    for (int s = 128; s > 0; s >>= 1) {
        if (d < s) { sf[d] += sf[d + s]; sd[d] += sd[d + s]; }
        __syncthreads();
    }
    if (d == 0) v[r] = v[r] - sf[0] * ht + sd[0];

    const float xn = xcur[off] - g * ht + noise;
    xcur[off] = xn;
    xbf[off]  = f2bf(xn);
}

// ---------------------------------------------------------------------------
// Pack outputs: d_out[0..B) = v ; d_out[B .. B+B*D) = xT
// ---------------------------------------------------------------------------
__global__ void pack_out_kernel(const float* __restrict__ v,
                                const float* __restrict__ xcur,
                                float* __restrict__ out, int total)
{
    int idx = blockIdx.x * blockDim.x + threadIdx.x;
    if (idx >= total) return;
    out[idx] = (idx < B_ROWS) ? v[idx] : xcur[idx - B_ROWS];
}

// ---------------------------------------------------------------------------
// Host-side orchestration
// ---------------------------------------------------------------------------
extern "C" void kernel_launch(void* const* d_in, const int* in_sizes, int n_in,
                              void* d_out, int out_size, void* d_ws, size_t ws_size,
                              hipStream_t stream)
{
    (void)in_sizes; (void)n_in; (void)out_size; (void)ws_size;

    const float* x   = (const float*)d_in[0];
    const float* xi  = (const float*)d_in[1];
    const float* tg  = (const float*)d_in[2];
    const float* W1  = (const float*)d_in[3];
    const float* b1  = (const float*)d_in[4];
    const float* g1  = (const float*)d_in[5];
    const float* be1 = (const float*)d_in[6];
    const float* W2  = (const float*)d_in[7];
    const float* b2  = (const float*)d_in[8];
    const float* g2  = (const float*)d_in[9];
    const float* be2 = (const float*)d_in[10];
    const float* W3  = (const float*)d_in[11];
    const float* b3  = (const float*)d_in[12];
    const float* vW1 = (const float*)d_in[13];
    const float* vb1 = (const float*)d_in[14];
    const float* vg1 = (const float*)d_in[15];
    const float* vbe1= (const float*)d_in[16];
    const float* vW2 = (const float*)d_in[17];
    const float* vb2 = (const float*)d_in[18];
    const float* vg2 = (const float*)d_in[19];
    const float* vbe2= (const float*)d_in[20];
    const float* vW3 = (const float*)d_in[21];
    const float* vb3 = (const float*)d_in[22];
    const float* vg3 = (const float*)d_in[23];
    const float* vbe3= (const float*)d_in[24];

    // ---- workspace carve (256B aligned blocks) ----
    char* p = (char*)d_ws;
    auto carve = [&](size_t bytes) -> void* {
        void* q = (void*)p;
        p += (bytes + 255) & ~(size_t)255;
        return q;
    };
    __bf16* W1t  = (__bf16*)carve((size_t)TSTEPS * 256 * HPAD_N * 2);
    __bf16* W2t  = (__bf16*)carve((size_t)TSTEPS * HPAD_K * HPAD_N * 2);
    __bf16* W3t  = (__bf16*)carve((size_t)TSTEPS * HPAD_K * 256 * 2);
    __bf16* vW1t = (__bf16*)carve((size_t)256 * HPAD_N * 2);
    __bf16* vW2t = (__bf16*)carve((size_t)HPAD_K * HPAD_N * 2);
    __bf16* vW3t = (__bf16*)carve((size_t)HPAD_K * 16 * 2);
    float*  xcur = (float*) carve((size_t)B_ROWS * DDIM * 4);
    __bf16* xbf  = (__bf16*)carve((size_t)B_ROWS * DDIM * 2);
    float*  ybuf = (float*) carve((size_t)B_ROWS * HPAD_N * 4);
    __bf16* hbf  = (__bf16*)carve((size_t)B_ROWS * HPAD_K * 2);
    float*  gradb= (float*) carve((size_t)B_ROWS * DDIM * 4);
    float*  vbuf = (float*) carve((size_t)B_ROWS * 4);
    float*  scale= (float*) carve((size_t)HPAD_K * 4);
    float*  shift= (float*) carve((size_t)HPAD_K * 4);

    auto cdiv = [](long long a, long long b) { return (int)((a + b - 1) / b); };

    // ---- weight prep ----
    {
        long long n;
        n = (long long)TSTEPS * 256 * HPAD_N;
        convert_w_kernel<<<cdiv(n,256), 256, 0, stream>>>(W1,  W1t,  HDIM, DDIM, HPAD_N, 256,    TSTEPS);
        n = (long long)TSTEPS * HPAD_K * HPAD_N;
        convert_w_kernel<<<cdiv(n,256), 256, 0, stream>>>(W2,  W2t,  HDIM, HDIM, HPAD_N, HPAD_K, TSTEPS);
        n = (long long)TSTEPS * HPAD_K * 256;
        convert_w_kernel<<<cdiv(n,256), 256, 0, stream>>>(W3,  W3t,  DDIM, HDIM, 256,    HPAD_K, TSTEPS);
        n = (long long)256 * HPAD_N;
        convert_w_kernel<<<cdiv(n,256), 256, 0, stream>>>(vW1, vW1t, HDIM, DDIM, HPAD_N, 256,    1);
        n = (long long)HPAD_K * HPAD_N;
        convert_w_kernel<<<cdiv(n,256), 256, 0, stream>>>(vW2, vW2t, HDIM, HDIM, HPAD_N, HPAD_K, 1);
        n = (long long)HPAD_K * 16;
        convert_w_kernel<<<cdiv(n,256), 256, 0, stream>>>(vW3, vW3t, 1,    HDIM, 16,     HPAD_K, 1);
    }

    init_x_kernel<<<(B_ROWS * DDIM) / 256, 256, 0, stream>>>(x, xcur, xbf);

    auto gemm = [&](const __bf16* A, int lda, const __bf16* Bt, int ldb,
                    const float* bias, int Nact, float* out, int ldo,
                    int Ntiles, int kChunks) {
        dim3 grid(B_ROWS / 64, cdiv(Ntiles, NT));
        gemm_bf16_kernel<<<grid, 128, 0, stream>>>(A, lda, Bt, ldb, bias, Nact,
                                                   out, ldo, kChunks, Ntiles);
    };
    auto stats = [&](const float* g, const float* be, int Hact, int nblk) {
        stats_kernel<<<nblk, dim3(32, 8), 0, stream>>>(ybuf, HPAD_N, B_ROWS,
                                                       g, be, Hact, scale, shift);
    };
    auto bnrelu = [&]() {
        bnrelu_kernel<<<(B_ROWS * (HPAD_K / 4)) / 256, 256, 0, stream>>>(
            ybuf, HPAD_N, scale, shift, hbf);
    };

    // ---- v0 sub-network ----
    gemm(xbf, DDIM,   vW1t, HPAD_N, vb1, HDIM, ybuf, HPAD_N, HPAD_N / 16, 256 / 32);
    stats(vg1, vbe1, HDIM, HPAD_K / 32);
    bnrelu();
    gemm(hbf, HPAD_K, vW2t, HPAD_N, vb2, HDIM, ybuf, HPAD_N, HPAD_N / 16, HPAD_K / 32);
    stats(vg2, vbe2, HDIM, HPAD_K / 32);
    bnrelu();
    gemm(hbf, HPAD_K, vW3t, 16,     vb3, 1,    ybuf, HPAD_N, 1,           HPAD_K / 32);
    stats(vg3, vbe3, 1, 1);
    v0_finalize_kernel<<<B_ROWS / 256, 256, 0, stream>>>(ybuf, HPAD_N, scale, shift, vbuf);

    // ---- time-step scan ----
    for (int t = 0; t < TSTEPS; ++t) {
        const __bf16* W1tt = W1t + (size_t)t * 256 * HPAD_N;
        const __bf16* W2tt = W2t + (size_t)t * HPAD_K * HPAD_N;
        const __bf16* W3tt = W3t + (size_t)t * HPAD_K * 256;

        gemm(xbf, DDIM,   W1tt, HPAD_N, b1 + (size_t)t * HDIM, HDIM,
             ybuf, HPAD_N, HPAD_N / 16, 256 / 32);
        stats(g1 + (size_t)t * HDIM, be1 + (size_t)t * HDIM, HDIM, HPAD_K / 32);
        bnrelu();

        gemm(hbf, HPAD_K, W2tt, HPAD_N, b2 + (size_t)t * HDIM, HDIM,
             ybuf, HPAD_N, HPAD_N / 16, HPAD_K / 32);
        stats(g2 + (size_t)t * HDIM, be2 + (size_t)t * HDIM, HDIM, HPAD_K / 32);
        bnrelu();

        gemm(hbf, HPAD_K, W3tt, 256,    b3 + (size_t)t * DDIM, DDIM,
             gradb, DDIM, DDIM / 16, HPAD_K / 32);

        update_kernel<<<B_ROWS, 256, 0, stream>>>(xcur, xbf, vbuf, gradb,
                                                  xi + (size_t)t * B_ROWS * DDIM,
                                                  tg, t);
    }

    // ---- pack outputs: (vT [B,1], xT [B,D]) ----
    const int total = B_ROWS + B_ROWS * DDIM;
    pack_out_kernel<<<cdiv(total, 256), 256, 0, stream>>>(vbuf, xcur,
                                                          (float*)d_out, total);
}
